// MLA_9268539425530
// MI455X (gfx1250) — matrix operand order repaired and checked by reading, not compiled, not run on hardware
//
#include <hip/hip_runtime.h>
#include <hip/hip_bf16.h>
#include <math.h>

// Problem constants (match reference)
#define Bz   2
#define Tz   2048
#define Cz   2048
#define NHz  16
#define DKz  128
#define DHRz 64
#define LATz 512
#define Mz   (Bz * Tz)   // 4096 flattened rows

typedef __attribute__((ext_vector_type(16))) __bf16 v16bf;
typedef __attribute__((ext_vector_type(8)))  float  v8f;
typedef __attribute__((ext_vector_type(4)))  unsigned int v4u;
typedef __attribute__((ext_vector_type(8)))  int v8i;
typedef __attribute__((ext_vector_type(4)))  int v4i;

union Frag16 {
    uint4 q[2];   // two 16B chunks = 16 bf16
    v16bf v;
};

static __device__ __forceinline__ unsigned short f2bf(float f) {
    union { float f; unsigned int u; } x; x.f = f;
    unsigned int u = x.u;
    unsigned int r = (u + 0x7FFFu + ((u >> 16) & 1u)) >> 16;  // round-to-nearest-even
    return (unsigned short)r;
}

static __device__ __forceinline__ v8f wmma_bf16(v16bf a, v16bf b, v8f c) {
    // D = A(16x32 bf16) * B(32x16 bf16) + C(16x16 f32)
    return __builtin_amdgcn_wmma_f32_16x16x32_bf16(false, a, false, b, (short)0, c, false, false);
}

// ---------------------------------------------------------------------------
// Tensor Data Mover: async 2D tile load Global -> LDS (CDNA5 TDM, D# per ISA 8.3/8.4).
// Loads tile_h rows x tile_w elements (2-byte elements) from a row-major tensor with
// row stride `stride_e` elements into contiguous LDS at byte offset lds_off.
// This toolchain exposes the 6-arg builtin: (v4u g0, v8i g1, v4i g2, v4i g3, v8i, i32 cpol).
static __device__ __forceinline__ void tdm_load_2d_bf16(unsigned lds_off,
                                                        const void* gptr,
                                                        unsigned tile_w, unsigned tile_h,
                                                        unsigned stride_e,
                                                        unsigned tensor_w, unsigned tensor_h) {
    unsigned long long ga = (unsigned long long)(size_t)gptr;
    v4u g0;
    g0.x = 1u;                                        // count=1 (valid user descriptor)
    g0.y = lds_off;                                   // LDS byte address
    g0.z = (unsigned)(ga & 0xFFFFFFFFull);            // global_addr[31:0]
    g0.w = (unsigned)((ga >> 32) & 0x1FFFFFFull)      // global_addr[56:32]
         | (2u << 30);                                // type = 2 ("image")
    v8i g1;
    g1[0] = (int)(1u << 16);                          // workgroup_mask=0, data_size=1 (2B)
    g1[1] = (int)((tensor_w & 0xFFFFu) << 16);        // abar_addr=0 | tensor_dim0[15:0]
    g1[2] = (int)((tensor_w >> 16) | ((tensor_h & 0xFFFFu) << 16)); // dim0[31:16] | dim1[15:0]
    g1[3] = (int)((tensor_h >> 16) | (tile_w << 16)); // dim1[31:16] | tile_dim0
    g1[4] = (int)tile_h;                              // tile_dim1 (tile_dim2 = 0)
    g1[5] = (int)stride_e;                            // tensor_dim0_stride[31:0]
    g1[6] = 0;                                        // stride[47:32]=0 | dim1_stride lo=0
    g1[7] = 0;
    v4i z4 = {0, 0, 0, 0};
    v8i z8 = {0, 0, 0, 0, 0, 0, 0, 0};
    __builtin_amdgcn_tensor_load_to_lds(g0, g1, z4, z4, z8, 0);
}

// ---------------------------------------------------------------------------
// Elementwise: f32 -> bf16 copy
__global__ void cvt_bf16_k(const float* __restrict__ src, unsigned short* __restrict__ dst, int n) {
    int i = blockIdx.x * 256 + threadIdx.x;
    if (i < n) dst[i] = f2bf(src[i]);
}

// Weight transpose: W (K x N) f32 -> Wt (N x K) bf16
__global__ void wtrans_k(const float* __restrict__ W, unsigned short* __restrict__ Wt, int K, int N) {
    int i = blockIdx.x * 256 + threadIdx.x;
    if (i >= K * N) return;
    int k = i / N, n = i % N;
    Wt[(size_t)n * K + k] = f2bf(W[i]);
}

// ---------------------------------------------------------------------------
// Small-N bf16 WMMA GEMM (used for the N=64 kR projection). One wave = 16x64 C tile.
template <int OUTF32>
__global__ void __launch_bounds__(256)
gemm_small_k(const unsigned short* __restrict__ A, int lda,
             const unsigned short* __restrict__ Wt, int K,
             const float* __restrict__ bias,
             void* __restrict__ Cp, int ldc,
             int tiles, int tilesN)
{
    const int lane = threadIdx.x & 31;
    const int wid  = threadIdx.x >> 5;
    const int l16  = lane & 15;
    const int hi   = lane >> 4;
    int gw = blockIdx.x * 8 + wid;
    if (gw >= tiles) return;
    const int mt = gw / tilesN;
    const int nt = gw % tilesN;

    const v8f zero8 = {0.f,0.f,0.f,0.f,0.f,0.f,0.f,0.f};
    v8f acc[4];
#pragma unroll
    for (int s = 0; s < 4; ++s) acc[s] = zero8;

    const unsigned short* arow = A + ((size_t)(mt * 16 + l16)) * lda;
    for (int k = 0; k < K; k += 32) {
        Frag16 af;
        af.q[0] = *reinterpret_cast<const uint4*>(arow + k + hi * 16);
        af.q[1] = *reinterpret_cast<const uint4*>(arow + k + hi * 16 + 8);
#pragma unroll
        for (int s = 0; s < 4; ++s) {
            Frag16 bfr;
            const unsigned short* wrow =
                Wt + ((size_t)(nt * 64 + s * 16 + l16)) * K + k + hi * 16;
            bfr.q[0] = *reinterpret_cast<const uint4*>(wrow);
            bfr.q[1] = *reinterpret_cast<const uint4*>(wrow + 8);
            acc[s] = wmma_bf16(af.v, bfr.v, acc[s]);
        }
    }
#pragma unroll
    for (int s = 0; s < 4; ++s) {
        const int col = nt * 64 + s * 16 + l16;
        const float bv = bias[col];
#pragma unroll
        for (int r = 0; r < 8; ++r) {
            const int row = mt * 16 + hi * 8 + r;
            float v = acc[s][r] + bv;
            if (OUTF32) ((float*)Cp)[(size_t)row * ldc + col] = v;
            else        ((unsigned short*)Cp)[(size_t)row * ldc + col] = f2bf(v);
        }
    }
}

// ---------------------------------------------------------------------------
// TDM-staged bf16 WMMA GEMM. Block (8 waves) computes a 64x256 C tile; the shared
// 64x32 A tile is DMA'd into double-buffered LDS by the Tensor Data Mover, while
// each wave streams its own 32 columns of Wt from global. Per k-step a wave does
// 8 WMMAs against 4 LDS A-fragments and 2 global B-fragments.
// Requires: M % 64 == 0, N % 256 == 0, K % 32 == 0.
#define TDM_TILE_BYTES (64 * 32 * 2)   // 4 KB per buffer
template <int OUTF32>
__global__ void __launch_bounds__(256)
gemm_tdm_k(const unsigned short* __restrict__ A, int lda,
           const unsigned short* __restrict__ Wt, int K,
           const float* __restrict__ bias,
           void* __restrict__ Cp, int ldc,
           int tilesN)
{
    __shared__ __align__(16) unsigned short smem[2 * 64 * 32];  // 8 KB, double buffered

    const int lane = threadIdx.x & 31;
    const int wid  = threadIdx.x >> 5;
    const int l16  = lane & 15;
    const int hi   = lane >> 4;
    const int mt = blockIdx.x / tilesN;
    const int nt = blockIdx.x % tilesN;
    const int mrow = mt * 64;

    const v8f zero8 = {0.f,0.f,0.f,0.f,0.f,0.f,0.f,0.f};
    v8f acc[8];   // [mi*2 + s] : mi = 16-row subtile (0..3), s = 16-col subtile (0..1)
#pragma unroll
    for (int i = 0; i < 8; ++i) acc[i] = zero8;

    const int ksteps = K / 32;

    // Preload tile 0 into buffer 0 (wave 0 only; TDM ignores EXEC, one op per wave).
    if (wid == 0) {
        tdm_load_2d_bf16(0u, A + (size_t)mrow * lda, 32u, 64u,
                         (unsigned)lda, (unsigned)K, (unsigned)Mz);
    }

    // B column pointers for this wave's two 16-col subtiles
    const unsigned short* wrow0 =
        Wt + ((size_t)(nt * 256 + wid * 32 + l16)) * K + hi * 16;
    const unsigned short* wrow1 =
        Wt + ((size_t)(nt * 256 + wid * 32 + 16 + l16)) * K + hi * 16;

    for (int i = 0; i < ksteps; ++i) {
        if (wid == 0) __builtin_amdgcn_s_wait_tensorcnt(0);  // tile i landed in LDS
        __syncthreads();  // tile i visible to all waves; parity (i+1)&1 readers retired
        if (wid == 0 && i + 1 < ksteps) {
            tdm_load_2d_bf16((unsigned)(((i + 1) & 1) * TDM_TILE_BYTES),
                             A + (size_t)mrow * lda + (size_t)(i + 1) * 32,
                             32u, 64u, (unsigned)lda, (unsigned)K, (unsigned)Mz);
        }
        const unsigned short* lbuf = smem + ((i & 1) ? (64 * 32) : 0);

        Frag16 af[4];
#pragma unroll
        for (int mi = 0; mi < 4; ++mi) {
            const unsigned short* arow = lbuf + (mi * 16 + l16) * 32;
            af[mi].q[0] = *reinterpret_cast<const uint4*>(arow + hi * 16);
            af[mi].q[1] = *reinterpret_cast<const uint4*>(arow + hi * 16 + 8);
        }
        {
            Frag16 b0, b1;
            b0.q[0] = *reinterpret_cast<const uint4*>(wrow0 + i * 32);
            b0.q[1] = *reinterpret_cast<const uint4*>(wrow0 + i * 32 + 8);
            b1.q[0] = *reinterpret_cast<const uint4*>(wrow1 + i * 32);
            b1.q[1] = *reinterpret_cast<const uint4*>(wrow1 + i * 32 + 8);
#pragma unroll
            for (int mi = 0; mi < 4; ++mi) {
                acc[mi * 2 + 0] = wmma_bf16(af[mi].v, b0.v, acc[mi * 2 + 0]);
                acc[mi * 2 + 1] = wmma_bf16(af[mi].v, b1.v, acc[mi * 2 + 1]);
            }
        }
    }

#pragma unroll
    for (int s = 0; s < 2; ++s) {
        const int col = nt * 256 + wid * 32 + s * 16 + l16;
        const float bv = bias[col];
#pragma unroll
        for (int mi = 0; mi < 4; ++mi) {
#pragma unroll
            for (int r = 0; r < 8; ++r) {
                const int row = mrow + mi * 16 + hi * 8 + r;   // C layout: M = r + 8*hi
                float v = acc[mi * 2 + s][r] + bv;
                if (OUTF32) ((float*)Cp)[(size_t)row * ldc + col] = v;
                else        ((unsigned short*)Cp)[(size_t)row * ldc + col] = f2bf(v);
            }
        }
    }
}

// ---------------------------------------------------------------------------
// RoPE (fp32 math, bf16 out). Reference: freq = t+1; theta_i = exp(-2*(i/2)/d * ln 1e4);
// out_even = x0*c - x1*s ; out_odd = x1*c + x0*s
#define LN1E4 9.210340371976184f

// kR: (B*T, 64) f32 -> (B, T, 64) bf16
__global__ void rope_k_k(const float* __restrict__ lin, unsigned short* __restrict__ out) {
    int i = blockIdx.x * 256 + threadIdx.x;   // pair index, total Mz * 32
    if (i >= Mz * (DHRz / 2)) return;
    int j   = i % (DHRz / 2);
    int row = i / (DHRz / 2);
    int t   = row % Tz;
    float theta = __expf(-2.0f * (float)j / (float)DHRz * LN1E4);
    float ang = (float)(t + 1) * theta;
    float c = cosf(ang), s = sinf(ang);
    float x0 = lin[(size_t)row * DHRz + 2 * j];
    float x1 = lin[(size_t)row * DHRz + 2 * j + 1];
    out[(size_t)row * DHRz + 2 * j]     = f2bf(x0 * c - x1 * s);
    out[(size_t)row * DHRz + 2 * j + 1] = f2bf(x1 * c + x0 * s);
}

// qR: (B*T, 1024) f32 -> (B, NH, T, 64) bf16 ; theta over full d = NH*DHR = 1024
__global__ void rope_q_k(const float* __restrict__ lin, unsigned short* __restrict__ out) {
    const int D = NHz * DHRz;                  // 1024
    int i = blockIdx.x * 256 + threadIdx.x;    // pair index, total Mz * 512
    if (i >= Mz * (D / 2)) return;
    int j   = i % (D / 2);
    int row = i / (D / 2);
    int t   = row % Tz;
    int b   = row / Tz;
    float theta = __expf(-2.0f * (float)j / (float)D * LN1E4);
    float ang = (float)(t + 1) * theta;
    float c = cosf(ang), s = sinf(ang);
    float x0 = lin[(size_t)row * D + 2 * j];
    float x1 = lin[(size_t)row * D + 2 * j + 1];
    int i0 = 2 * j;
    int h  = i0 >> 6;          // /64 ; pair never straddles a head (64 even)
    int d0 = i0 & 63;
    size_t base = (((size_t)b * NHz + h) * Tz + t) * DHRz;
    out[base + d0]     = f2bf(x0 * c - x1 * s);
    out[base + d0 + 1] = f2bf(x1 * c + x0 * s);
}

// V transpose: kv (B*T, 4096) bf16 (second half = V) -> Vt (B, NH, DK, T) bf16
__global__ void vtrans_k(const unsigned short* __restrict__ kv, unsigned short* __restrict__ vt) {
    int i = blockIdx.x * 256 + threadIdx.x;   // over B*NH*DK*T, t fastest (coalesced writes)
    if (i >= Bz * NHz * DKz * Tz) return;
    int t = i % Tz; int rest = i / Tz;
    int d = rest % DKz; rest /= DKz;
    int h = rest % NHz;
    int b = rest / NHz;
    vt[i] = kv[((size_t)(b * Tz) + t) * (2 * Cz) + Cz + h * DKz + d];
}

// ---------------------------------------------------------------------------
// Flash-style causal attention. One wave per (b, h, 16-query tile).
// scores = q·k (128 dims, from kv buffer) + qR·kR (64 dims). scale = 1/sqrt(128).
__global__ void __launch_bounds__(128)
attn_k(const unsigned short* __restrict__ qn,   // (B*T, 2048) bf16
       const unsigned short* __restrict__ qr,   // (B, NH, T, 64) bf16
       const unsigned short* __restrict__ kv,   // (B*T, 4096) bf16 (first half = K)
       const unsigned short* __restrict__ kr,   // (B*T, 64) bf16
       const unsigned short* __restrict__ vt,   // (B, NH, DK, T) bf16
       unsigned short* __restrict__ attn)       // (B*T, 2048) bf16
{
    __shared__ __align__(16) unsigned short lds[4 * 512];   // per-wave 16x32 P tile
    const int lane = threadIdx.x & 31;
    const int wid  = threadIdx.x >> 5;
    const int l16  = lane & 15;
    const int hi   = lane >> 4;
    int gw = blockIdx.x * 4 + wid;
    const int QT = Tz / 16;
    if (gw >= Bz * NHz * QT) return;
    const int qt = gw % QT;
    const int h  = (gw / QT) % NHz;
    const int b  = gw / (QT * NHz);
    const int qbase = qt * 16;
    unsigned short* pl = &lds[wid * 512];

    // Q fragments: 4 chunks from the no-pe part, 2 chunks from the rope part (192 dims total)
    Frag16 qf[6];
    {
        const unsigned short* qrow = qn + ((size_t)(b * Tz + qbase + l16)) * Cz + h * DKz;
#pragma unroll
        for (int c = 0; c < 4; ++c) {
            qf[c].q[0] = *reinterpret_cast<const uint4*>(qrow + c * 32 + hi * 16);
            qf[c].q[1] = *reinterpret_cast<const uint4*>(qrow + c * 32 + hi * 16 + 8);
        }
        const unsigned short* qrrow = qr + (((size_t)(b * NHz + h)) * Tz + qbase + l16) * DHRz;
#pragma unroll
        for (int c = 0; c < 2; ++c) {
            qf[4 + c].q[0] = *reinterpret_cast<const uint4*>(qrrow + c * 32 + hi * 16);
            qf[4 + c].q[1] = *reinterpret_cast<const uint4*>(qrrow + c * 32 + hi * 16 + 8);
        }
    }

    const v8f zero8 = {0.f,0.f,0.f,0.f,0.f,0.f,0.f,0.f};
    v8f acc[8];
#pragma unroll
    for (int d = 0; d < 8; ++d) acc[d] = zero8;
    float m[8], lsum[8];
#pragma unroll
    for (int r = 0; r < 8; ++r) { m[r] = -1e30f; lsum[r] = 0.f; }

    const int nkb = (qbase + 15) / 32 + 1;        // causal: only key blocks <= query tile
    const float scale = 0.08838834764831845f;     // 1/sqrt(128)

    for (int j = 0; j < nkb; ++j) {
        const int kb = j * 32;
        v8f s0 = zero8, s1 = zero8;               // scores, keys [kb..kb+15], [kb+16..kb+31]

        const unsigned short* k0p = kv + ((size_t)(b * Tz + kb + l16)) * (2 * Cz) + h * DKz;
        const unsigned short* k1p = kv + ((size_t)(b * Tz + kb + 16 + l16)) * (2 * Cz) + h * DKz;
#pragma unroll
        for (int c = 0; c < 4; ++c) {
            Frag16 b0, b1;
            b0.q[0] = *reinterpret_cast<const uint4*>(k0p + c * 32 + hi * 16);
            b0.q[1] = *reinterpret_cast<const uint4*>(k0p + c * 32 + hi * 16 + 8);
            b1.q[0] = *reinterpret_cast<const uint4*>(k1p + c * 32 + hi * 16);
            b1.q[1] = *reinterpret_cast<const uint4*>(k1p + c * 32 + hi * 16 + 8);
            s0 = wmma_bf16(qf[c].v, b0.v, s0);
            s1 = wmma_bf16(qf[c].v, b1.v, s1);
        }
        const unsigned short* kr0 = kr + ((size_t)(b * Tz + kb + l16)) * DHRz;
        const unsigned short* kr1 = kr + ((size_t)(b * Tz + kb + 16 + l16)) * DHRz;
#pragma unroll
        for (int c = 0; c < 2; ++c) {
            Frag16 b0, b1;
            b0.q[0] = *reinterpret_cast<const uint4*>(kr0 + c * 32 + hi * 16);
            b0.q[1] = *reinterpret_cast<const uint4*>(kr0 + c * 32 + hi * 16 + 8);
            b1.q[0] = *reinterpret_cast<const uint4*>(kr1 + c * 32 + hi * 16);
            b1.q[1] = *reinterpret_cast<const uint4*>(kr1 + c * 32 + hi * 16 + 8);
            s0 = wmma_bf16(qf[4 + c].v, b0.v, s0);
            s1 = wmma_bf16(qf[4 + c].v, b1.v, s1);
        }

        // scale + causal mask (C layout: row = qbase + r + 8*hi, col = l16 [+16 for s1])
        const int tq = qbase + hi * 8;
#pragma unroll
        for (int r = 0; r < 8; ++r) {
            float v0 = s0[r] * scale; if (kb + l16 > tq + r)      v0 = -1e30f;
            float v1 = s1[r] * scale; if (kb + 16 + l16 > tq + r) v1 = -1e30f;
            s0[r] = v0; s1[r] = v1;
        }

        // row max across 32 keys: reduce within each 16-lane half (rows 0-7 / 8-15)
        float rm[8];
#pragma unroll
        for (int r = 0; r < 8; ++r) rm[r] = fmaxf(s0[r], s1[r]);
#pragma unroll
        for (int msk = 8; msk >= 1; msk >>= 1) {
#pragma unroll
            for (int r = 0; r < 8; ++r) rm[r] = fmaxf(rm[r], __shfl_xor(rm[r], msk, 32));
        }
        float alpha[8];
#pragma unroll
        for (int r = 0; r < 8; ++r) {
            float mn = fmaxf(m[r], rm[r]);
            alpha[r] = __expf(m[r] - mn);
            m[r] = mn;
        }
#pragma unroll
        for (int d = 0; d < 8; ++d)
#pragma unroll
            for (int r = 0; r < 8; ++r) acc[d][r] *= alpha[r];

        float rs[8];
#pragma unroll
        for (int r = 0; r < 8; ++r) {
            s0[r] = __expf(s0[r] - m[r]);
            s1[r] = __expf(s1[r] - m[r]);
            rs[r] = s0[r] + s1[r];
        }
#pragma unroll
        for (int msk = 8; msk >= 1; msk >>= 1) {
#pragma unroll
            for (int r = 0; r < 8; ++r) rs[r] += __shfl_xor(rs[r], msk, 32);
        }
#pragma unroll
        for (int r = 0; r < 8; ++r) lsum[r] = lsum[r] * alpha[r] + rs[r];

        // P: C-layout fp32 -> A-layout bf16 via LDS (DS ops are in-order within a wave)
#pragma unroll
        for (int r = 0; r < 8; ++r) {
            pl[(hi * 8 + r) * 32 + l16]      = f2bf(s0[r]);
            pl[(hi * 8 + r) * 32 + 16 + l16] = f2bf(s1[r]);
        }
        Frag16 pf;
        pf.q[0] = *reinterpret_cast<const uint4*>(pl + l16 * 32 + hi * 16);
        pf.q[1] = *reinterpret_cast<const uint4*>(pl + l16 * 32 + hi * 16 + 8);

        // O += P(16x32) * V(32x16) per 16-wide output dim tile
#pragma unroll
        for (int dt = 0; dt < 8; ++dt) {
            Frag16 vf;
            const unsigned short* vrow =
                vt + (((size_t)(b * NHz + h)) * DKz + dt * 16 + l16) * Tz + kb + hi * 16;
            vf.q[0] = *reinterpret_cast<const uint4*>(vrow);
            vf.q[1] = *reinterpret_cast<const uint4*>(vrow + 8);
            acc[dt] = wmma_bf16(pf.v, vf.v, acc[dt]);
        }
    }

    float inv[8];
#pragma unroll
    for (int r = 0; r < 8; ++r) inv[r] = 1.0f / lsum[r];
#pragma unroll
    for (int dt = 0; dt < 8; ++dt)
#pragma unroll
        for (int r = 0; r < 8; ++r)
            attn[((size_t)(b * Tz + qbase + hi * 8 + r)) * Cz + h * DKz + dt * 16 + l16] =
                f2bf(acc[dt][r] * inv[r]);
}

// ---------------------------------------------------------------------------
extern "C" void kernel_launch(void* const* d_in, const int* in_sizes, int n_in,
                              void* d_out, int out_size, void* d_ws, size_t ws_size,
                              hipStream_t stream) {
    const float* x      = (const float*)d_in[0];
    const float* Wdown  = (const float*)d_in[1];
    const float* bdown  = (const float*)d_in[2];
    const float* WuKV   = (const float*)d_in[3];
    const float* buKV   = (const float*)d_in[4];
    const float* WuQ    = (const float*)d_in[5];
    const float* buQ    = (const float*)d_in[6];
    const float* Wkr    = (const float*)d_in[7];
    const float* bkr    = (const float*)d_in[8];
    const float* Wqr    = (const float*)d_in[9];
    const float* bqr    = (const float*)d_in[10];
    const float* Wout   = (const float*)d_in[11];
    const float* bout   = (const float*)d_in[12];
    float* out = (float*)d_out;

    // carve workspace
    char* wsp = (char*)d_ws;
    auto carve = [&](size_t bytes) -> char* {
        char* p = wsp;
        wsp += (bytes + 255) & ~(size_t)255;
        return p;
    };
    unsigned short* x_bf      = (unsigned short*)carve(2ull * Mz * Cz);
    unsigned short* wdown_t   = (unsigned short*)carve(2ull * 1024 * 2048);
    unsigned short* wukv_t    = (unsigned short*)carve(2ull * 4096 * 512);
    unsigned short* wuq_t     = (unsigned short*)carve(2ull * 2048 * 512);
    unsigned short* wkr_t     = (unsigned short*)carve(2ull * 64 * 2048);
    unsigned short* wqr_t     = (unsigned short*)carve(2ull * 1024 * 2048);
    unsigned short* wout_t    = (unsigned short*)carve(2ull * 2048 * 2048);
    unsigned short* latent_bf = (unsigned short*)carve(2ull * Mz * 1024);
    unsigned short* q_bf      = (unsigned short*)carve(2ull * Mz * 2048);
    unsigned short* kv_bf     = (unsigned short*)carve(2ull * Mz * 4096);
    unsigned short* kr_bf     = (unsigned short*)carve(2ull * Mz * DHRz);
    unsigned short* qr_bf     = (unsigned short*)carve(2ull * Mz * NHz * DHRz);
    unsigned short* vt_bf     = (unsigned short*)carve(2ull * Bz * NHz * DKz * Tz);
    unsigned short* attn_bf   = (unsigned short*)carve(2ull * Mz * Cz);
    float*          krlin     = (float*)carve(4ull * Mz * DHRz);
    float*          qrlin     = (float*)carve(4ull * Mz * NHz * DHRz);

    auto eblocks = [](int n) { return (n + 255) / 256; };

    // 1) convert activations + weights
    cvt_bf16_k<<<eblocks(Mz * Cz), 256, 0, stream>>>(x, x_bf, Mz * Cz);
    wtrans_k<<<eblocks(2048 * 1024), 256, 0, stream>>>(Wdown, wdown_t, 2048, 1024);
    wtrans_k<<<eblocks(512 * 4096),  256, 0, stream>>>(WuKV,  wukv_t,  512,  4096);
    wtrans_k<<<eblocks(512 * 2048),  256, 0, stream>>>(WuQ,   wuq_t,   512,  2048);
    wtrans_k<<<eblocks(2048 * 64),   256, 0, stream>>>(Wkr,   wkr_t,   2048, 64);
    wtrans_k<<<eblocks(2048 * 1024), 256, 0, stream>>>(Wqr,   wqr_t,   2048, 1024);
    wtrans_k<<<eblocks(2048 * 2048), 256, 0, stream>>>(Wout,  wout_t,  2048, 2048);

    // TDM-staged GEMMs (N % 256 == 0)
    auto gemm_f32 = [&](const unsigned short* A, int lda, const unsigned short* Wt, int K,
                        const float* bias, float* C, int ldc, int N) {
        int tilesN = N / 256;
        gemm_tdm_k<1><<<(Mz / 64) * tilesN, 256, 0, stream>>>(A, lda, Wt, K, bias, (void*)C,
                                                              ldc, tilesN);
    };
    auto gemm_bf = [&](const unsigned short* A, int lda, const unsigned short* Wt, int K,
                       const float* bias, unsigned short* C, int ldc, int N) {
        int tilesN = N / 256;
        gemm_tdm_k<0><<<(Mz / 64) * tilesN, 256, 0, stream>>>(A, lda, Wt, K, bias, (void*)C,
                                                              ldc, tilesN);
    };

    // 2) projections from x
    gemm_bf (x_bf, Cz, wdown_t, Cz, bdown, latent_bf, 1024, 1024);   // latent (cKV|cQ)
    gemm_f32(x_bf, Cz, wqr_t,   Cz, bqr,   qrlin,     1024, 1024);   // qR pre-rope
    {   // kR pre-rope: N=64 -> small per-wave GEMM
        int tilesN = 64 / 64;
        int tiles  = (Mz / 16) * tilesN;
        gemm_small_k<1><<<(tiles + 7) / 8, 256, 0, stream>>>(x_bf, Cz, wkr_t, Cz, bkr,
                                                             (void*)krlin, 64, tiles, tilesN);
    }

    // 3) RoPE
    rope_k_k<<<eblocks(Mz * (DHRz / 2)), 256, 0, stream>>>(krlin, kr_bf);
    rope_q_k<<<eblocks(Mz * (NHz * DHRz / 2)), 256, 0, stream>>>(qrlin, qr_bf);

    // 4) up projections from latent halves
    gemm_bf(latent_bf,       1024, wukv_t, LATz, buKV, kv_bf, 4096, 4096);  // k|v
    gemm_bf(latent_bf + 512, 1024, wuq_t,  LATz, buQ,  q_bf,  2048, 2048);  // q

    // 5) V transpose for PV WMMA
    vtrans_k<<<eblocks(Bz * NHz * DKz * Tz), 256, 0, stream>>>(kv_bf, vt_bf);

    // 6) causal attention (one wave per 16-query tile per head)
    {
        int waves  = Bz * NHz * (Tz / 16);   // 4096
        int blocks = (waves + 3) / 4;
        attn_k<<<blocks, 128, 0, stream>>>(q_bf, qr_bf, kv_bf, kr_bf, vt_bf, attn_bf);
    }

    // 7) output projection -> fp32 result
    gemm_f32(attn_bf, Cz, wout_t, Cz, bout, out, Cz, 2048);
    (void)in_sizes; (void)n_in; (void)out_size; (void)ws_size;
}
